// GraphConvClf_3607772528843
// MI455X (gfx1250) — matrix operand order repaired and checked by reading, not compiled, not run on hardware
//
#include <hip/hip_runtime.h>
#include <hip/hip_bf16.h>

typedef __attribute__((ext_vector_type(16))) __bf16 v16bf;
typedef __attribute__((ext_vector_type(8)))  float  v8f;

#define DIM 128
#define WSTRIDE 134   // LDS row stride in bf16 elems (= 67 dwords, odd -> spread banks)

// ---------------------------------------------------------------------------
// Dual GEMM: agg = x @ w0^T + b0 ; x1out = x @ w1^T + b1  (optionally relu(x))
// One wave computes a 16-row tile against all 8 column tiles of both weights.
// bf16 WMMA 16x16x32, fp32 accumulation.
// ---------------------------------------------------------------------------
__global__ __launch_bounds__(256) void gconv_dual_gemm(
    const float* __restrict__ x,
    const float* __restrict__ w0, const float* __restrict__ b0,
    const float* __restrict__ w1, const float* __restrict__ b1,
    float* __restrict__ agg, float* __restrict__ x1out,
    int nrows, int apply_relu)
{
  __shared__ __bf16 w0s[DIM * WSTRIDE];
  __shared__ __bf16 w1s[DIM * WSTRIDE];
  const int tid = threadIdx.x;
  for (int idx = tid; idx < DIM * DIM; idx += 256) {
    const int r = idx >> 7, c = idx & 127;
    w0s[r * WSTRIDE + c] = (__bf16)w0[idx];
    w1s[r * WSTRIDE + c] = (__bf16)w1[idx];
  }
  __syncthreads();

  const int wave = tid >> 5;
  const int lane = tid & 31;
  const int m    = lane & 15;
  const int half = lane >> 4;

  const long long row0 = ((long long)blockIdx.x * 8 + wave) * 16;
  if (row0 + 16 > nrows) return;

  // ---- A fragments: 16x128 row tile as 4 k-steps of 16x32 bf16 ----
  // 16-bit A layout: lanes 0-15 M=lane, V0:K=0,1 .. V3:K=6,7 (lanes 16-31: +8),
  // V4..V7: K=16..23 / 24..31.
  v16bf afrag[4];
  const float* xrow = x + (row0 + m) * DIM;
  #pragma unroll
  for (int s = 0; s < 4; ++s) {
    #pragma unroll
    for (int v = 0; v < 8; ++v) {
      const int k = s * 32 + ((v < 4) ? (2 * v + 8 * half)
                                      : (16 + 2 * (v - 4) + 8 * half));
      float f0 = xrow[k];
      float f1 = xrow[k + 1];
      if (apply_relu) { f0 = fmaxf(f0, 0.f); f1 = fmaxf(f1, 0.f); }
      afrag[s][2 * v]     = (__bf16)f0;
      afrag[s][2 * v + 1] = (__bf16)f1;
    }
  }

  #pragma unroll
  for (int ht = 0; ht < 8; ++ht) {
    const int n = ht * 16 + m;          // output column handled by this lane
    v8f acc0 = {};
    v8f acc1 = {};
    #pragma unroll
    for (int s = 0; s < 4; ++s) {
      // B layout (K x 16): same K order as A: V0..V3 K=0..7 (lanes 16-31: +8), V4..V7 K=16..23 / 24..31.
      // B[k][n] = w[n][k]  (out = x @ w^T)
      v16bf bf0, bf1;
      #pragma unroll
      for (int v = 0; v < 8; ++v) {
        const int k = s * 32 + ((v < 4) ? (2 * v + 8 * half) : (16 + 2 * (v - 4) + 8 * half));
        bf0[2 * v]     = w0s[n * WSTRIDE + k];
        bf0[2 * v + 1] = w0s[n * WSTRIDE + k + 1];
        bf1[2 * v]     = w1s[n * WSTRIDE + k];
        bf1[2 * v + 1] = w1s[n * WSTRIDE + k + 1];
      }
      acc0 = __builtin_amdgcn_wmma_f32_16x16x32_bf16(false, afrag[s], false, bf0,
                                                     (short)0, acc0, false, false);
      acc1 = __builtin_amdgcn_wmma_f32_16x16x32_bf16(false, afrag[s], false, bf1,
                                                     (short)0, acc1, false, false);
    }
    const float bias0 = b0[n];
    const float bias1 = b1[n];
    // D layout: lanes 0-15 -> M=r, lanes 16-31 -> M=r+8, N=lane&15
    #pragma unroll
    for (int r = 0; r < 8; ++r) {
      const long long orow = row0 + r + half * 8;
      agg[orow * DIM + n]   = acc0[r] + bias0;
      x1out[orow * DIM + n] = acc1[r] + bias1;
    }
  }
}

// ---------------------------------------------------------------------------
// Edge scatter: agg[i] += x1[j]; agg[j] += x1[i].  One wave per edge,
// 4 channels per lane, f32 global atomics.
// ---------------------------------------------------------------------------
__global__ __launch_bounds__(256) void gconv_scatter(
    const long long* __restrict__ edges,
    const float* __restrict__ x1,
    float* __restrict__ agg, long long nedges)
{
  const long long e = (long long)blockIdx.x * 8 + (threadIdx.x >> 5);
  if (e >= nedges) return;
  const int lane = threadIdx.x & 31;
  const long long i = edges[2 * e];
  const long long j = edges[2 * e + 1];
  const float4 vi = ((const float4*)(x1 + i * DIM))[lane];
  const float4 vj = ((const float4*)(x1 + j * DIM))[lane];
  float* ai = agg + i * DIM + lane * 4;
  float* aj = agg + j * DIM + lane * 4;
  atomicAdd(ai + 0, vj.x); atomicAdd(ai + 1, vj.y);
  atomicAdd(ai + 2, vj.z); atomicAdd(ai + 3, vj.w);
  atomicAdd(aj + 0, vi.x); atomicAdd(aj + 1, vi.y);
  atomicAdd(aj + 2, vi.z); atomicAdd(aj + 3, vi.w);
}

// ---------------------------------------------------------------------------
// Mean-pool over V vertices per mesh, with final-layer ReLU folded in.
// ---------------------------------------------------------------------------
__global__ __launch_bounds__(256) void pool_mean_relu(
    const float* __restrict__ x, float* __restrict__ pooled, int V)
{
  const int mesh = blockIdx.x;
  const int c = threadIdx.x & 127;
  const int g = threadIdx.x >> 7;   // 0 or 1
  const float* base = x + (long long)mesh * V * DIM;
  float acc = 0.f;
  for (int r = g; r < V; r += 2)
    acc += fmaxf(base[(long long)r * DIM + c], 0.f);
  __shared__ float red[256];
  red[threadIdx.x] = acc;
  __syncthreads();
  if (g == 0)
    pooled[mesh * DIM + c] = (red[c] + red[128 + c]) / (float)V;
}

// ---------------------------------------------------------------------------
// h = relu(pooled @ fc1_w^T + fc1_b)   (16 x 1024, thread per output)
// ---------------------------------------------------------------------------
__global__ __launch_bounds__(256) void fc1_relu(
    const float* __restrict__ pooled, const float* __restrict__ w,
    const float* __restrict__ b, float* __restrict__ h)
{
  const int idx  = blockIdx.x * 256 + threadIdx.x;   // 0 .. 16383
  const int mesh = idx >> 10;
  const int o    = idx & 1023;
  const float* p  = pooled + mesh * DIM;
  const float* wr = w + o * DIM;
  float acc = b[o];
  #pragma unroll 4
  for (int k = 0; k < DIM; ++k) acc += p[k] * wr[k];
  h[idx] = fmaxf(acc, 0.f);
}

// ---------------------------------------------------------------------------
// Four classification heads, concatenated flat in return order (240 floats).
// ---------------------------------------------------------------------------
__global__ __launch_bounds__(256) void head_gemms(
    const float* __restrict__ h,
    const float* __restrict__ w_st, const float* __restrict__ b_st,
    const float* __restrict__ w_se, const float* __restrict__ b_se,
    const float* __restrict__ w_fu, const float* __restrict__ b_fu,
    const float* __restrict__ w_ae, const float* __restrict__ b_ae,
    float* __restrict__ out)
{
  const int t = blockIdx.x * blockDim.x + threadIdx.x;
  if (t >= 240) return;
  const float* w; const float* b; int C; int local;
  if (t < 48)       { w = w_st; b = b_st; C = 3; local = t; }        // style 16x3
  else if (t < 80)  { w = w_se; b = b_se; C = 2; local = t - 48; }   // semantics 16x2
  else if (t < 160) { w = w_fu; b = b_fu; C = 5; local = t - 80; }   // functionality 16x5
  else              { w = w_ae; b = b_ae; C = 5; local = t - 160; }  // aesthetics 16x5
  const int mesh = local / C, c = local % C;
  const float* hv = h + mesh * 1024;
  const float* wr = w + c * 1024;
  float acc = b[c];
  #pragma unroll 4
  for (int k = 0; k < 1024; ++k) acc += hv[k] * wr[k];
  out[t] = acc;
}

// ---------------------------------------------------------------------------
extern "C" void kernel_launch(void* const* d_in, const int* in_sizes, int n_in,
                              void* d_out, int out_size, void* d_ws, size_t ws_size,
                              hipStream_t stream) {
  const float*     verts = (const float*)d_in[0];
  const long long* edges = (const long long*)d_in[1];   // int64 pairs, batch-offset baked in
  const float* w0[3] = {(const float*)d_in[2], (const float*)d_in[6],  (const float*)d_in[10]};
  const float* b0[3] = {(const float*)d_in[3], (const float*)d_in[7],  (const float*)d_in[11]};
  const float* w1[3] = {(const float*)d_in[4], (const float*)d_in[8],  (const float*)d_in[12]};
  const float* b1[3] = {(const float*)d_in[5], (const float*)d_in[9],  (const float*)d_in[13]};
  const float* fc1_w = (const float*)d_in[14];
  const float* fc1_b = (const float*)d_in[15];
  const float* w_st = (const float*)d_in[16]; const float* b_st = (const float*)d_in[17];
  const float* w_se = (const float*)d_in[18]; const float* b_se = (const float*)d_in[19];
  const float* w_fu = (const float*)d_in[20]; const float* b_fu = (const float*)d_in[21];
  const float* w_ae = (const float*)d_in[22]; const float* b_ae = (const float*)d_in[23];

  const int       nrows  = in_sizes[0] / DIM;        // 400000
  const long long nedges = in_sizes[1] / 2;          // 1,200,000
  const int B = 16;
  const int V = nrows / B;

  // Workspace: bufA, bufB (x/agg ping-pong), bufX1, pooled, h
  const size_t nxh = (size_t)nrows * DIM;
  float* bufA   = (float*)d_ws;
  float* bufB   = bufA + nxh;
  float* bufX1  = bufB + nxh;
  float* pooled = bufX1 + nxh;
  float* hbuf   = pooled + (size_t)B * DIM;

  const dim3 blk(256);
  const int gemm_blocks = (nrows + 127) / 128;                 // 8 waves x 16 rows
  const int scat_blocks = (int)((nedges + 7) / 8);             // 8 edges / block

  const float* xin = verts;
  float* aggs[3] = {bufA, bufB, bufA};
  for (int l = 0; l < 3; ++l) {
    gconv_dual_gemm<<<gemm_blocks, blk, 0, stream>>>(
        xin, w0[l], b0[l], w1[l], b1[l], aggs[l], bufX1, nrows, (l > 0) ? 1 : 0);
    gconv_scatter<<<scat_blocks, blk, 0, stream>>>(edges, bufX1, aggs[l], nedges);
    xin = aggs[l];   // relu folded into next consumer
  }
  pool_mean_relu<<<B, blk, 0, stream>>>(aggs[2], pooled, V);
  fc1_relu<<<(B * 1024) / 256, blk, 0, stream>>>(pooled, fc1_w, fc1_b, hbuf);
  head_gemms<<<1, blk, 0, stream>>>(hbuf, w_st, b_st, w_se, b_se,
                                    w_fu, b_fu, w_ae, b_ae, (float*)d_out);
}
